// Heterogeneus_3659312136870
// MI455X (gfx1250) — matrix operand order repaired and checked by reading, not compiled, not run on hardware
//
#include <hip/hip_runtime.h>
#include <hip/hip_bf16.h>

// ---------------------------------------------------------------------------
// Hetero-GNN forward for MI455X (gfx1250), wave32 + v_wmma_f32_16x16x32_f16.
//   N=100000 nodes/type, E=800000 edges/rel, F0=64, H=128, G=64, L=7.
// Scatter (fp32 atomics, BW-bound) + WMMA-fused GEMMs (f16 in, f32 accum).
// GEMM: block = 8 waves = 32 rows x 128 cols; A and B fragments both feed
// from LDS as aligned ds_load_b128 (sW stored transposed / column-major).
// ---------------------------------------------------------------------------

typedef __attribute__((ext_vector_type(16))) _Float16 v16h;
typedef __attribute__((ext_vector_type(8)))  _Float16 h8;
typedef __attribute__((ext_vector_type(4)))  _Float16 h4;
typedef __attribute__((ext_vector_type(8)))  float    v8f;

#define GN      100000
#define GE      800000
#define GH      128
#define GG      64
#define HC      384
#define TILE_K  128
#define LDSS    (TILE_K + 8)    // padded LDS row stride (halves), 272B: 16B-aligned

// ------------------------------- utility -----------------------------------
__global__ void zero_f32(float* __restrict__ p, long n) {
  long i = (long)blockIdx.x * blockDim.x + threadIdx.x;
  long stride = (long)gridDim.x * blockDim.x;
  for (; i < n; i += stride) p[i] = 0.0f;
}

// --------------------------- edge scatter-sum -------------------------------
// agg[dst, :] += x[src, :] over all edges.  F in {64,128}, F % 4 == 0.
__global__ void scatter_add(const int* __restrict__ eSrc,
                            const int* __restrict__ eDst,
                            const float* __restrict__ x,
                            float* __restrict__ agg, int F) {
  const int chunks = F >> 2;
  const long total = (long)GE * chunks;
  long i = (long)blockIdx.x * blockDim.x + threadIdx.x;
  long stride = (long)gridDim.x * blockDim.x;
  for (; i < total; i += stride) {
    int e = (int)(i / chunks);
    int c = (int)(i % chunks) << 2;
    int s = eSrc[e], d = eDst[e];
    const float4 v = *(const float4*)(x + (size_t)s * F + c);
    float* o = agg + (size_t)d * F + c;
    atomicAdd(o + 0, v.x); atomicAdd(o + 1, v.y);
    atomicAdd(o + 2, v.z); atomicAdd(o + 3, v.w);
  }
}

// ------------------------- fused multi-GEMM (WMMA) ---------------------------
// out[M, Hout] = relu?( sum_p A_p[M,K] @ W_p[K,Hout] + sum_p bias_p )
// Block = 8 waves: rows [32*bx, 32*bx+32), cols [128*by, 128*by+128).
struct GemmArgs {
  const float* A[6];
  const float* W[6];
  const float* bias[6];
  int nmat, M, K, Hout, relu;
  float* out;
};

__global__ void __launch_bounds__(256)
gemm_fused(GemmArgs args) {
  __shared__ _Float16 sA[32 * LDSS];       // 8.5 KB, row-major  [m][k]
  __shared__ _Float16 sW[128 * LDSS];      // 34 KB,  col-major  [n][k]
  const int tid  = threadIdx.x;
  const int wave = tid >> 5;               // 0..7 -> 16-col tile
  const int lane = tid & 31;
  const int rowBase = blockIdx.x * 32;
  const int colBase = blockIdx.y * 128;

  v8f c0 = {}, c1 = {};                    // fp32 accumulators, two row tiles

  const int mrow = lane & 15;
  const int kbA  = (lane >> 4) << 4;       // A frag: lane-group K base (0 / 16)
  const int kbB  = (lane >> 4) << 4;       // B frag: lane-group K base (0 / 16)
  const int ncol = wave * 16 + (lane & 15);

  for (int p = 0; p < args.nmat; ++p) {
    const float* __restrict__ A = args.A[p];
    const float* __restrict__ W = args.W[p];
    for (int k0 = 0; k0 < args.K; k0 += TILE_K) {
      // stage A tile (32 x TILE_K) as f16; float4 reads, 4-half (b64) stores
      for (int cid = tid; cid < 32 * (TILE_K / 4); cid += 256) {
        int r  = cid >> 5;                 // 0..31
        int c4 = (cid & 31) << 2;          // 0..124
        int gr = rowBase + r, gk = k0 + c4;
        float4 v = make_float4(0.f, 0.f, 0.f, 0.f);
        if (gr < args.M && gk < args.K)
          v = *(const float4*)(A + (size_t)gr * args.K + gk);
        h4 hv = { (_Float16)v.x, (_Float16)v.y, (_Float16)v.z, (_Float16)v.w };
        *(h4*)(sA + r * LDSS + c4) = hv;
      }
      // stage W tile transposed: sW[n][k] = W[k0+k][colBase+n]; coalesced reads
      for (int cid = tid; cid < TILE_K * (128 / 4); cid += 256) {
        int k  = cid >> 5;                 // 0..127
        int n4 = (cid & 31) << 2;          // 0..124
        int gk = k0 + k;
        float4 v = make_float4(0.f, 0.f, 0.f, 0.f);
        if (gk < args.K)
          v = *(const float4*)(W + (size_t)gk * args.Hout + colBase + n4);
        sW[(n4 + 0) * LDSS + k] = (_Float16)v.x;
        sW[(n4 + 1) * LDSS + k] = (_Float16)v.y;
        sW[(n4 + 2) * LDSS + k] = (_Float16)v.z;
        sW[(n4 + 3) * LDSS + k] = (_Float16)v.w;
      }
      if (k0 + TILE_K < args.K)   // hint next W strip toward L2 (global_prefetch_b8)
        __builtin_prefetch(W + (size_t)(k0 + TILE_K) * args.Hout + colBase + tid, 0, 1);
      __syncthreads();

      #pragma unroll
      for (int kk = 0; kk < TILE_K; kk += 32) {
        // B fragment: column ncol, K = kk+kbB .. +15 contiguous -> 2x b128
        const _Float16* pB = sW + ncol * LDSS + kk + kbB;
        h8 blo = *(const h8*)(pB);
        h8 bhi = *(const h8*)(pB + 8);
        v16h b = __builtin_shufflevector(blo, bhi,
                   0,1,2,3,4,5,6,7,8,9,10,11,12,13,14,15);
        // A fragments (two row tiles): contiguous run [kb..kb+15]
        const _Float16* pA0 = sA + mrow * LDSS + kk + kbA;
        h8 a0lo = *(const h8*)(pA0);
        h8 a0hi = *(const h8*)(pA0 + 8);
        v16h a0 = __builtin_shufflevector(a0lo, a0hi,
                    0,1,2,3,4,5,6,7,8,9,10,11,12,13,14,15);
        const _Float16* pA1 = sA + (16 + mrow) * LDSS + kk + kbA;
        h8 a1lo = *(const h8*)(pA1);
        h8 a1hi = *(const h8*)(pA1 + 8);
        v16h a1 = __builtin_shufflevector(a1lo, a1hi,
                    0,1,2,3,4,5,6,7,8,9,10,11,12,13,14,15);

        c0 = __builtin_amdgcn_wmma_f32_16x16x32_f16(
                 false, a0, false, b, (short)0, c0, false, false);
        c1 = __builtin_amdgcn_wmma_f32_16x16x32_f16(
                 false, a1, false, b, (short)0, c1, false, false);
      }
      __syncthreads();
    }
  }

  // bias + relu + store (C/D layout: vgpr j -> M=j (+8 for lanes 16-31))
  const int gcol = colBase + ncol;
  float bsum = 0.0f;
  for (int p = 0; p < args.nmat; ++p)
    if (args.bias[p]) bsum += args.bias[p][gcol];
  const int moff = (lane >> 4) << 3;
  #pragma unroll
  for (int j = 0; j < 8; ++j) {
    int g0 = rowBase + j + moff;
    if (g0 < args.M) {
      float v = c0[j] + bsum;
      if (args.relu) v = fmaxf(v, 0.0f);
      args.out[(size_t)g0 * args.Hout + gcol] = v;
    }
    int g1 = rowBase + 16 + j + moff;
    if (g1 < args.M) {
      float v = c1[j] + bsum;
      if (args.relu) v = fmaxf(v, 0.0f);
      args.out[(size_t)g1 * args.Hout + gcol] = v;
    }
  }
}

// ------------------------------- pooling ------------------------------------
__global__ void pool_sum(const float* __restrict__ x, const int* __restrict__ batch,
                         float* __restrict__ h, float* __restrict__ cnt,
                         int colOff, int typeIdx) {
  const long total = (long)GN * 32;        // 128/4 chunks per node
  long i = (long)blockIdx.x * blockDim.x + threadIdx.x;
  long stride = (long)gridDim.x * blockDim.x;
  for (; i < total; i += stride) {
    int nidx = (int)(i >> 5);
    int c = (int)(i & 31) << 2;
    int g = batch[nidx];
    const float4 v = *(const float4*)(x + (size_t)nidx * GH + c);
    float* o = h + (size_t)g * HC + colOff + c;
    atomicAdd(o + 0, v.x); atomicAdd(o + 1, v.y);
    atomicAdd(o + 2, v.z); atomicAdd(o + 3, v.w);
    if (c == 0) atomicAdd(cnt + typeIdx * GG + g, 1.0f);
  }
}

__global__ void pool_div(float* __restrict__ h, const float* __restrict__ cnt) {
  int i = blockIdx.x * blockDim.x + threadIdx.x;
  if (i < GG * HC) {
    int g = i / HC, col = i % HC, t = col >> 7;
    h[i] /= fmaxf(cnt[t * GG + g], 1.0f);
  }
}

// ------------------------------ output head ---------------------------------
__global__ void head_out(const float* __restrict__ h, const float* __restrict__ Wout,
                         const float* __restrict__ bout, float* __restrict__ out) {
  int g = threadIdx.x;
  if (g < GG) {
    float s = bout[0];
    for (int j = 0; j < HC; ++j) s += h[g * HC + j] * Wout[j];
    out[g] = s;
  }
}

// ------------------------------- host side ----------------------------------
extern "C" void kernel_launch(void* const* d_in, const int* in_sizes, int n_in,
                              void* d_out, int out_size, void* d_ws, size_t ws_size,
                              hipStream_t stream) {
  (void)in_sizes; (void)n_in; (void)out_size; (void)ws_size;
  const float* x_in[3] = { (const float*)d_in[0], (const float*)d_in[1], (const float*)d_in[2] };
  const int* e_a1_cols = (const int*)d_in[3];
  const int* e_a0_b    = (const int*)d_in[4];
  const int* e_a1_b    = (const int*)d_in[5];
  const int* e_b_b     = (const int*)d_in[6];
  const int* e_a0_rows = (const int*)d_in[7];
  const int* e_a1_rows = (const int*)d_in[8];
  const int* batch[3]  = { (const int*)d_in[9], (const int*)d_in[10], (const int*)d_in[11] };

  // params flattened jax-style (sorted dict keys), starting at d_in[13]:
  //  layers[l] x rel-sorted {a0_b,a0_rows,a1_b,a1_cols,a1_rows,b_b} x {Wr,Wt,br}
  auto P = [&](int i) { return (const float*)d_in[13 + i]; };
  enum { R_A0B = 0, R_A0ROWS = 1, R_A1B = 2, R_A1COLS = 3, R_A1ROWS = 4, R_BB = 5 };
  const float* lin1W = P(126), *lin1b = P(127);
  const float* lin2W = P(128), *lin2b = P(129);
  const float* lin3W = P(130), *lin3b = P(131);
  const float* outW  = P(132), *outb  = P(133);

  // workspace layout (fp32):
  float* ws = (float*)d_ws;
  const size_t NH = (size_t)GN * GH;
  float* xb[2][3];
  for (int s = 0; s < 2; ++s)
    for (int t = 0; t < 3; ++t) xb[s][t] = ws + ((size_t)(s * 3 + t)) * NH;
  float* agg[3] = { ws + 6 * NH, ws + 7 * NH, ws + 8 * NH };
  float* hbuf = ws + 9 * NH;                // [G, 384]
  float* cnt  = hbuf + (size_t)GG * HC;     // [3, G]
  float* h2   = cnt + 3 * GG;               // [G, 384]
  float* h3   = h2 + (size_t)GG * HC;       // [G, 384]

  auto zero = [&](float* p, long n) {
    int blocks = (int)((n + 255) / 256); if (blocks > 65535) blocks = 65535;
    zero_f32<<<blocks, 256, 0, stream>>>(p, n);
  };
  auto scatter = [&](const int* edges, const float* x, float* a, int F) {
    long total = (long)GE * (F >> 2);
    int blocks = (int)((total + 255) / 256); if (blocks > 65535) blocks = 65535;
    scatter_add<<<blocks, 256, 0, stream>>>(edges, edges + GE, x, a, F);
  };
  auto gemm = [&](int nmat, const float* const* As, const float* const* Ws,
                  const float* const* Bs, int M, int K, int Hout, int relu, float* out) {
    GemmArgs ga{};
    ga.nmat = nmat; ga.M = M; ga.K = K; ga.Hout = Hout; ga.relu = relu; ga.out = out;
    for (int i = 0; i < nmat; ++i) { ga.A[i] = As[i]; ga.W[i] = Ws[i]; ga.bias[i] = Bs[i]; }
    dim3 grid((M + 31) / 32, Hout / 128);
    gemm_fused<<<grid, 256, 0, stream>>>(ga);
  };

  const float* cx[3] = { x_in[0], x_in[1], x_in[2] };
  int K = 64;
  for (int l = 0; l < 7; ++l) {
    auto Wr = [&](int r) { return P(l * 18 + r * 3 + 0); };
    auto Wt = [&](int r) { return P(l * 18 + r * 3 + 1); };
    auto br = [&](int r) { return P(l * 18 + r * 3 + 2); };
    float* nx[3] = { xb[l & 1][0], xb[l & 1][1], xb[l & 1][2] };
    const size_t NK = (size_t)GN * K;

    // dst a_0 : rel a1_rows (a_1 -> a_0)
    zero(agg[0], (long)NK);
    scatter(e_a1_rows, cx[1], agg[0], K);
    { const float* As[2] = { agg[0], cx[0] };
      const float* Ws[2] = { Wr(R_A1ROWS), Wt(R_A1ROWS) };
      const float* Bs[2] = { br(R_A1ROWS), nullptr };
      gemm(2, As, Ws, Bs, GN, K, GH, 1, nx[0]); }

    // dst a_1 : rels a1_cols (a_1->a_1) + a0_rows (a_0->a_1)
    zero(agg[0], (long)NK); zero(agg[1], (long)NK);
    scatter(e_a1_cols, cx[1], agg[0], K);
    scatter(e_a0_rows, cx[0], agg[1], K);
    { const float* As[4] = { agg[0], cx[1], agg[1], cx[1] };
      const float* Ws[4] = { Wr(R_A1COLS), Wt(R_A1COLS), Wr(R_A0ROWS), Wt(R_A0ROWS) };
      const float* Bs[4] = { br(R_A1COLS), nullptr, br(R_A0ROWS), nullptr };
      gemm(4, As, Ws, Bs, GN, K, GH, 1, nx[1]); }

    // dst b : rels a0_b + a1_b + b_b
    zero(agg[0], (long)NK); zero(agg[1], (long)NK); zero(agg[2], (long)NK);
    scatter(e_a0_b, cx[0], agg[0], K);
    scatter(e_a1_b, cx[1], agg[1], K);
    scatter(e_b_b,  cx[2], agg[2], K);
    { const float* As[6] = { agg[0], cx[2], agg[1], cx[2], agg[2], cx[2] };
      const float* Ws[6] = { Wr(R_A0B), Wt(R_A0B), Wr(R_A1B), Wt(R_A1B), Wr(R_BB), Wt(R_BB) };
      const float* Bs[6] = { br(R_A0B), nullptr, br(R_A1B), nullptr, br(R_BB), nullptr };
      gemm(6, As, Ws, Bs, GN, K, GH, 1, nx[2]); }

    cx[0] = nx[0]; cx[1] = nx[1]; cx[2] = nx[2];
    K = GH;
  }

  // mean pooling into hbuf[G, 384]
  zero(hbuf, (long)(GG * HC + 3 * GG));
  pool_sum<<<12500, 256, 0, stream>>>(cx[0], batch[0], hbuf, cnt, 0,   0);
  pool_sum<<<12500, 256, 0, stream>>>(cx[1], batch[1], hbuf, cnt, 128, 1);
  pool_sum<<<12500, 256, 0, stream>>>(cx[2], batch[2], hbuf, cnt, 256, 2);
  pool_div<<<(GG * HC + 255) / 256, 256, 0, stream>>>(hbuf, cnt);

  // MLP head: 3 x (384x384) relu GEMMs on WMMA, then dot with out weights
  { const float* As[1] = { hbuf }; const float* Ws[1] = { lin1W }; const float* Bs[1] = { lin1b };
    gemm(1, As, Ws, Bs, GG, HC, HC, 1, h2); }
  { const float* As[1] = { h2 };   const float* Ws[1] = { lin2W }; const float* Bs[1] = { lin2b };
    gemm(1, As, Ws, Bs, GG, HC, HC, 1, h3); }
  { const float* As[1] = { h3 };   const float* Ws[1] = { lin3W }; const float* Bs[1] = { lin3b };
    gemm(1, As, Ws, Bs, GG, HC, HC, 1, h2); }
  head_out<<<1, 64, 0, stream>>>(h2, outW, outb, (float*)d_out);
}
